// FlashAttention_33380485824832
// MI455X (gfx1250) — matrix operand order repaired and checked by reading, not compiled, not run on hardware
//
#include <hip/hip_runtime.h>
#include <hip/hip_bf16.h>

typedef __attribute__((ext_vector_type(16))) _Float16 v16h;
typedef __attribute__((ext_vector_type(8)))  _Float16 v8h;
typedef __attribute__((ext_vector_type(8)))  float    v8f;

#define FA_NEG_INF (-1.0e30f)

// Flash attention fwd, causal + sliding window (W=512), GQA 16q/4kv heads.
// One block = 64 query rows of one (b,h); 4 waves (wave32), each wave owns a
// 16-row slice. Raw f32 K/V tiles (32 keys) are double-buffered into LDS via
// GLOBAL_LOAD_ASYNC_TO_LDS_B128 (ASYNCcnt pipelined one tile ahead), then
// converted to f16 staging tiles (V transposed) feeding v_wmma_f32_16x16x32_f16.
__launch_bounds__(128, 1)
__global__ void fa_fwd_kernel(const float* __restrict__ q,
                              const float* __restrict__ k,
                              const float* __restrict__ v,
                              float* __restrict__ out)
{
    constexpr int Hq = 16, Hkv = 4, S = 2048, D = 128, W = 512;
    constexpr float scale = 0.08838834764831845f; // 1/sqrt(128)

    __shared__ alignas(16) float    Kraw[2][32 * D];  // raw f32 async landing pads
    __shared__ alignas(16) float    Vraw[2][32 * D];
    __shared__ alignas(32) _Float16 Ks[32 * D];       // [key][d] f16
    __shared__ alignas(32) _Float16 VsT[D * 32];      // [d][key] f16 (transposed)
    __shared__ alignas(32) _Float16 Ps[4][16 * 32];   // per-wave P scratch

    const int tid  = threadIdx.x;
    const int wave = tid >> 5;
    const int lane = tid & 31;
    const int r    = lane & 15;
    const int hf   = lane >> 4;

    const int b   = blockIdx.z;
    const int h   = blockIdx.y;
    const int m0  = blockIdx.x * 64;
    const int mw  = m0 + wave * 16;
    const int kvh = h >> 2;   // G = 4

    const float* qp = q   + ((size_t)(b * Hq  + h)   * S) * D;
    const float* kp = k   + ((size_t)(b * Hkv + kvh) * S) * D;
    const float* vp = v   + ((size_t)(b * Hkv + kvh) * S) * D;
    float*       op = out + ((size_t)(b * Hq  + h)   * S) * D;

    // LDS byte offsets (flat shared pointer low 32 bits = LDS offset)
    const unsigned kraw_base = (unsigned)(size_t)&Kraw[0][0];
    const unsigned vraw_base = (unsigned)(size_t)&Vraw[0][0];

    // ---- async prefetch of one 32-key raw K/V tile into buffer `buf` ----
    // 8 rounds x 128 lanes x 16B = 16KB per tensor; 16 async ops per wave.
    auto prefetch_tile = [&](int nbase, int buf) {
        #pragma unroll
        for (int rnd = 0; rnd < 8; ++rnd) {
            const int e   = tid + rnd * 128;
            const int key = e >> 5;        // 32 float4 per 128-float row
            const int d4  = e & 31;
            const unsigned loff = (unsigned)(key * D + d4 * 4) * 4u;
            const float* gk = kp + (size_t)(nbase + key) * D + d4 * 4;
            const float* gv = vp + (size_t)(nbase + key) * D + d4 * 4;
            const unsigned lk = kraw_base + (unsigned)buf * (32u * D * 4u) + loff;
            const unsigned lv = vraw_base + (unsigned)buf * (32u * D * 4u) + loff;
            asm volatile("global_load_async_to_lds_b128 %0, %1, off"
                         :: "v"(lk), "v"(gk) : "memory");
            asm volatile("global_load_async_to_lds_b128 %0, %1, off"
                         :: "v"(lv), "v"(gv) : "memory");
        }
    };

    // ---- Q A-fragments (16x32 f16 each, 4 chunks over D=128), pre-scaled ----
    v16h qf[4];
    {
        const float* qrow = qp + (size_t)(mw + r) * D;
        #pragma unroll
        for (int c = 0; c < 4; ++c) {
            const int d0 = c * 32 + hf * 16;
            const float* p0 = qrow + d0;
            const float* p1 = qrow + d0 + 8;
            v16h a;
            #pragma unroll
            for (int t = 0; t < 8; ++t) {
                a[t]     = (_Float16)(p0[t] * scale);
                a[8 + t] = (_Float16)(p1[t] * scale);
            }
            qf[c] = a;
        }
    }

    // ---- accumulators (C/D layout: element g -> row mw + g + 8*hf) ----
    float m_i[8], l_i[8];
    v8f   o[8];
    #pragma unroll
    for (int g = 0; g < 8; ++g) { m_i[g] = FA_NEG_INF; l_i[g] = 0.0f; }
    #pragma unroll
    for (int c = 0; c < 8; ++c) {
        #pragma unroll
        for (int g = 0; g < 8; ++g) o[c][g] = 0.0f;
    }

    // ---- key-tile range: union of the 4 waves' causal+window footprints ----
    int blo = m0 - (W - 1);
    if (blo < 0) blo = 0;
    blo &= ~31;
    const int bhi      = m0 + 63;  // last visible key (inclusive)
    const int last_tile = m0 + 32; // last valid tile start

    prefetch_tile(blo, 0);         // prologue: tile 0 in flight

    int bufc = 0;
    for (int nb = blo; nb <= bhi; nb += 32) {
        // issue next tile into the other buffer (clamped duplicate on last iter
        // so the wait threshold below stays a constant 16)
        int nn = nb + 32;
        if (nn > last_tile) nn = last_tile;
        prefetch_tile(nn, bufc ^ 1);

        // wait: async ops complete in order -> <=16 outstanding means the 16
        // copies of the *current* tile have landed in LDS
        asm volatile("s_wait_asynccnt 0x10" ::: "memory");
        __syncthreads();   // all waves' copies visible; prior readers done

        // ---- convert raw f32 tile -> f16 staging (V transposed) ----
        const float* kr = &Kraw[bufc][0];
        const float* vr = &Vraw[bufc][0];
        #pragma unroll 8
        for (int e = tid; e < 32 * D; e += 128) {
            const int key = e >> 7;
            const int d   = e & (D - 1);
            Ks[key * D + d]   = (_Float16)kr[key * D + d];
            VsT[d * 32 + key] = (_Float16)vr[key * D + d];
        }
        __syncthreads();

        // ---- S = (Q*scale) @ K^T : two 16-col groups, chained over D ----
        v8f sA, sB;
        #pragma unroll
        for (int g = 0; g < 8; ++g) { sA[g] = 0.0f; sB[g] = 0.0f; }
        #pragma unroll
        for (int c = 0; c < 4; ++c) {
            const v16h bA = *(const v16h*)(Ks + (r)      * D + c * 32 + hf * 16);
            const v16h bB = *(const v16h*)(Ks + (r + 16) * D + c * 32 + hf * 16);
            sA = __builtin_amdgcn_wmma_f32_16x16x32_f16(false, qf[c], false, bA,
                                                        (short)0, sA, false, false);
            sB = __builtin_amdgcn_wmma_f32_16x16x32_f16(false, qf[c], false, bB,
                                                        (short)0, sB, false, false);
        }

        // ---- mask + online softmax ----
        const int jA = nb + r;
        const int jB = nb + 16 + r;
        bool  kA[8], kB[8];
        float mt[8];
        #pragma unroll
        for (int g = 0; g < 8; ++g) {
            const int irow = mw + g + 8 * hf;
            kA[g] = (jA <= irow) && ((irow - jA) < W);
            kB[g] = (jB <= irow) && ((irow - jB) < W);
            const float a  = kA[g] ? sA[g] : FA_NEG_INF;
            const float bq = kB[g] ? sB[g] : FA_NEG_INF;
            mt[g] = fmaxf(a, bq);
        }
        #pragma unroll
        for (int off = 8; off >= 1; off >>= 1) {
            #pragma unroll
            for (int g = 0; g < 8; ++g)
                mt[g] = fmaxf(mt[g], __shfl_xor(mt[g], off, 32));
        }

        float pAv[8], pBv[8], rs[8], alpha[8];
        #pragma unroll
        for (int g = 0; g < 8; ++g) {
            const float mn = fmaxf(m_i[g], mt[g]);
            alpha[g] = __expf(m_i[g] - mn);
            pAv[g] = kA[g] ? __expf(sA[g] - mn) : 0.0f;  // explicit zero on mask:
            pBv[g] = kB[g] ? __expf(sB[g] - mn) : 0.0f;  // safe for fully-masked tiles
            rs[g]  = pAv[g] + pBv[g];
            m_i[g] = mn;
        }
        #pragma unroll
        for (int off = 8; off >= 1; off >>= 1) {
            #pragma unroll
            for (int g = 0; g < 8; ++g)
                rs[g] += __shfl_xor(rs[g], off, 32);
        }
        #pragma unroll
        for (int g = 0; g < 8; ++g) l_i[g] = l_i[g] * alpha[g] + rs[g];
        #pragma unroll
        for (int c = 0; c < 8; ++c) {
            #pragma unroll
            for (int g = 0; g < 8; ++g) o[c][g] *= alpha[g];
        }

        // ---- C-layout P -> A-layout P via wave-private LDS scratch ----
        _Float16* Pw = &Ps[wave][0];
        #pragma unroll
        for (int g = 0; g < 8; ++g) {
            const int rl = g + 8 * hf;
            Pw[rl * 32 + r]      = (_Float16)pAv[g];
            Pw[rl * 32 + 16 + r] = (_Float16)pBv[g];
        }
        __builtin_amdgcn_wave_barrier();   // same-wave DS ops are in-order in HW
        v16h pfrag;
        {
            const v8h lo = *(const v8h*)(Pw + r * 32 + hf * 16);
            const v8h hi = *(const v8h*)(Pw + r * 32 + hf * 16 + 8);
            pfrag = __builtin_shufflevector(lo, hi,
                     0,1,2,3,4,5,6,7,8,9,10,11,12,13,14,15);
        }

        // ---- O += P @ V : 8 d-column groups of 16, K-dim = 32 keys ----
        #pragma unroll
        for (int c = 0; c < 8; ++c) {
            const v16h bV = *(const v16h*)(VsT + (c * 16 + r) * 32 + hf * 16);
            o[c] = __builtin_amdgcn_wmma_f32_16x16x32_f16(false, pfrag, false, bV,
                                                          (short)0, o[c], false, false);
        }

        bufc ^= 1;
    }

    // ---- normalize and store (every row has >=1 valid key, l > 0) ----
    #pragma unroll
    for (int g = 0; g < 8; ++g) {
        const float inv = 1.0f / l_i[g];
        const int   row = mw + g + 8 * hf;
        #pragma unroll
        for (int c = 0; c < 8; ++c)
            op[(size_t)row * D + c * 16 + r] = o[c][g] * inv;
    }
}

extern "C" void kernel_launch(void* const* d_in, const int* in_sizes, int n_in,
                              void* d_out, int out_size, void* d_ws, size_t ws_size,
                              hipStream_t stream) {
    const float* q = (const float*)d_in[0];
    const float* k = (const float*)d_in[1];
    const float* v = (const float*)d_in[2];
    float* out = (float*)d_out;

    // B=2, H=16, S=2048 -> grid (S/64, H, B), 128 threads (4 wave32)
    dim3 grid(2048 / 64, 16, 2);
    dim3 block(128);
    fa_fwd_kernel<<<grid, block, 0, stream>>>(q, k, v, out);
}
